// TokenCompetition_11793980195382
// MI455X (gfx1250) — compile-verified, matrix operand order checked by reading, not run on hardware
//
#include <hip/hip_runtime.h>
#include <hip/hip_bf16.h>
#include <math.h>

// ---------------------------------------------------------------------------
// Problem constants (match reference: B=2048, T=4, D=256, H=4, NIMG=64)
// ---------------------------------------------------------------------------
#define BN   2048
#define TN   4
#define DN   256
#define HN   4
#define HD   64
#define NIMG 64

#define LDS_PITCH 272   // 64x256 bf16 W chunk, row pitch padded (bank spread)

typedef __attribute__((ext_vector_type(16))) __bf16 v16bf;
typedef __attribute__((ext_vector_type(8)))  float  v8f;

// ---------------------------------------------------------------------------
// K0: group construction. counts[g], offs[g], members (tokens sorted by img),
//     multi[b] = (group size > 1). Single block, deterministic.
// ---------------------------------------------------------------------------
__global__ void tc_build_groups(const int* __restrict__ ids,
                                int* __restrict__ counts,
                                int* __restrict__ offs,
                                int* __restrict__ members,
                                int* __restrict__ multi) {
    int tid = threadIdx.x;
    if (tid < NIMG) {
        int c = 0;
        for (int b = 0; b < BN; ++b) c += (ids[b] == tid);
        counts[tid] = c;
    }
    __syncthreads();
    if (tid == 0) {
        int s = 0;
        for (int g = 0; g < NIMG; ++g) { offs[g] = s; s += counts[g]; }
    }
    __syncthreads();
    if (tid < NIMG) {
        int w = offs[tid];
        for (int b = 0; b < BN; ++b)
            if (ids[b] == tid) members[w++] = b;
    }
    __syncthreads();
    for (int b = tid; b < BN; b += blockDim.x)
        multi[b] = (counts[ids[b]] > 1) ? 1 : 0;
}

// ---------------------------------------------------------------------------
// K1a: x [B,T,D] f32 -> xsb [T,B,D] bf16 and cc [T,B,2D] bf16 (first half)
// ---------------------------------------------------------------------------
__global__ void tc_convert_x(const float* __restrict__ x,
                             __bf16* __restrict__ xsb,
                             __bf16* __restrict__ cc) {
    long i = (long)blockIdx.x * blockDim.x + threadIdx.x;
    const long n = (long)TN * BN * DN;
    if (i >= n) return;
    int d   = (int)(i & (DN - 1));
    long bt = i >> 8;                 // / DN
    int b   = (int)(bt & (BN - 1));
    int t   = (int)(bt >> 11);        // / BN
    float v = x[((long)b * TN + t) * DN + d];
    __bf16 bv = (__bf16)v;
    xsb[i] = bv;
    cc[((long)t * BN + b) * (2 * DN) + d] = bv;
}

// K1b: flat f32 -> bf16
__global__ void tc_convert_flat(const float* __restrict__ s,
                                __bf16* __restrict__ d, long n) {
    long i = (long)blockIdx.x * blockDim.x + threadIdx.x;
    if (i < n) d[i] = (__bf16)s[i];
}

// ---------------------------------------------------------------------------
// WMMA fragment loader: 16 rows x 32 K of bf16, row-major, row pitch LD
// (compile-time). 16-bit A/B layout (ISA 7.12.2): lanes 0-15 hold
// K {0..7,16..23}, lanes 16-31 hold K {8..15,24..31}. Two contiguous
// 16-byte runs per lane -> global_load_b128 / ds_load_b128 pairs.
// ---------------------------------------------------------------------------
template <int LD>
__device__ inline v16bf tc_load_frag16x32(const __bf16* __restrict__ base) {
    int lane = threadIdx.x & 31;
    int r    = lane & 15;
    int ks   = (lane >> 4) << 3;          // 0 or 8
    const __bf16* p = base + r * LD + ks;
    v16bf f;
#pragma unroll
    for (int j = 0; j < 8; ++j) {
        f[j]     = p[j];                  // K = k0 + ks + j
        f[j + 8] = p[16 + j];             // K = k0 + 16 + ks + j
    }
    return f;
}

// ---------------------------------------------------------------------------
// K2/K4/K5: generic batched GEMM  C[bz] = A[bz] (MxK) * W[bz]^T (NxK) + bias,
// optional ReLU, optional bf16 mirror with its own row stride (for concat).
//
// Block = 256 threads (8 waves) computing a 256x64 C tile.
//  * The shared 64-row W chunk (64 x 256 K) is staged into LDS ONCE per block
//    with CDNA5 async copies (global_load_async_to_lds_b128, ASYNCcnt), so
//    8 waves read it via ds_load instead of 8x redundant L2 traffic.
//  * Each wave computes 32x64: 2 A-frags (global) x 4 B-frags (LDS) ->
//    8 WMMAs per K-step on independent accumulators (no D->A/B hazards).
// ---------------------------------------------------------------------------
template <int KV>
__global__ void __launch_bounds__(256)
tc_gemm_wmma_bf16(const __bf16* __restrict__ A, long sA,
                  const __bf16* __restrict__ W, long sW,
                  const float* __restrict__ bias, long sBias,
                  float* __restrict__ C, long sC,
                  __bf16* __restrict__ Cb, long sCb, int ldcb,
                  int M, int N, int relu) {
    __shared__ __bf16 sWt[64 * LDS_PITCH];

    int ntb = N >> 6;                       // 64-wide N blocks
    int mtb = M >> 8;                       // 256-row M blocks
    int per = mtb * ntb;
    int bz  = blockIdx.x / per;
    int r   = blockIdx.x - bz * per;
    int mib = r / ntb;
    int nb  = r - mib * ntb;

    int wave = threadIdx.x >> 5;
    int m0   = mib * 256 + wave * 32;

    const __bf16* Ab = A + bz * sA + (long)m0 * KV;
    const __bf16* Wb = W + bz * sW + (long)nb * 64 * KV;

    unsigned ldsBase = (unsigned)(unsigned long)(void*)&sWt[0];

    v8f acc[8];
#pragma unroll
    for (int i = 0; i < 8; ++i) acc[i] = (v8f){};

    for (int kc = 0; kc < KV; kc += 256) {
        // ---- async stage: 64 x 256 bf16 W chunk -> LDS (32 KB) ----
        {
            int row = threadIdx.x >> 2;     // 0..63
            int seg = threadIdx.x & 3;      // 0..3 (4 x 128B per row)
            const __bf16* src = Wb + (long)row * KV + kc + seg * 64;
            unsigned dst = ldsBase + (unsigned)(row * LDS_PITCH + seg * 64) * 2u;
#pragma unroll
            for (int i = 0; i < 8; ++i) {
                unsigned long ga = (unsigned long)(src + i * 8);
                unsigned la = dst + (unsigned)i * 16u;
                asm volatile("global_load_async_to_lds_b128 %0, %1, off"
                             :: "v"(la), "v"(ga) : "memory");
            }
            asm volatile("s_wait_asynccnt 0x0" ::: "memory");
        }
        __syncthreads();

        // ---- compute: 8 K-steps of 32 over this chunk ----
#pragma unroll
        for (int ks = 0; ks < 256; ks += 32) {
            v16bf a0 = tc_load_frag16x32<KV>(Ab + kc + ks);
            v16bf a1 = tc_load_frag16x32<KV>(Ab + 16 * KV + kc + ks);
            v16bf b0 = tc_load_frag16x32<LDS_PITCH>(sWt + 0  * 16 * LDS_PITCH + ks);
            v16bf b1 = tc_load_frag16x32<LDS_PITCH>(sWt + 1  * 16 * LDS_PITCH + ks);
            v16bf b2 = tc_load_frag16x32<LDS_PITCH>(sWt + 2  * 16 * LDS_PITCH + ks);
            v16bf b3 = tc_load_frag16x32<LDS_PITCH>(sWt + 3  * 16 * LDS_PITCH + ks);
            acc[0] = __builtin_amdgcn_wmma_f32_16x16x32_bf16(
                false, a0, false, b0, (short)0, acc[0], false, false);
            acc[1] = __builtin_amdgcn_wmma_f32_16x16x32_bf16(
                false, a0, false, b1, (short)0, acc[1], false, false);
            acc[2] = __builtin_amdgcn_wmma_f32_16x16x32_bf16(
                false, a0, false, b2, (short)0, acc[2], false, false);
            acc[3] = __builtin_amdgcn_wmma_f32_16x16x32_bf16(
                false, a0, false, b3, (short)0, acc[3], false, false);
            acc[4] = __builtin_amdgcn_wmma_f32_16x16x32_bf16(
                false, a1, false, b0, (short)0, acc[4], false, false);
            acc[5] = __builtin_amdgcn_wmma_f32_16x16x32_bf16(
                false, a1, false, b1, (short)0, acc[5], false, false);
            acc[6] = __builtin_amdgcn_wmma_f32_16x16x32_bf16(
                false, a1, false, b2, (short)0, acc[6], false, false);
            acc[7] = __builtin_amdgcn_wmma_f32_16x16x32_bf16(
                false, a1, false, b3, (short)0, acc[7], false, false);
        }
        __syncthreads();                    // before next chunk overwrites LDS
    }

    // ---- epilogue: bias + optional relu + store (2 M-tiles x 4 N-tiles) ----
    int lane = threadIdx.x & 31;
    int n    = lane & 15;
    int mb   = (lane >> 4) << 3;            // 0 or 8
#pragma unroll
    for (int mt = 0; mt < 2; ++mt) {
#pragma unroll
        for (int nt = 0; nt < 4; ++nt) {
            v8f a = acc[mt * 4 + nt];
            int ncol = nb * 64 + nt * 16 + n;
            float bv = bias ? bias[bz * sBias + ncol] : 0.f;
#pragma unroll
            for (int rr = 0; rr < 8; ++rr) {
                float v = a[rr] + bv;
                if (relu) v = fmaxf(v, 0.f);
                int row = m0 + mt * 16 + mb + rr;
                C[bz * sC + (long)row * N + ncol] = v;
                if (Cb) Cb[bz * sCb + (long)row * ldcb + ncol] = (__bf16)v;
            }
        }
    }
}

// ---------------------------------------------------------------------------
// K3: grouped attention. Block per (group g, slot t). Thread per (row, head).
// Exact same math as the masked dense softmax (masked lanes contribute 0).
// Writes attention output directly as bf16 [T,B,D] for the out_proj GEMM.
// ---------------------------------------------------------------------------
__global__ void tc_attention(const float* __restrict__ qkv,
                             const int* __restrict__ counts,
                             const int* __restrict__ offs,
                             const int* __restrict__ members,
                             __bf16* __restrict__ attn_bf) {
    int g = blockIdx.x & (NIMG - 1);
    int t = blockIdx.x >> 6;
    int G = counts[g];
    if (G <= 0) return;
    int off = offs[g];
    int ntask = G * HN;

    for (int task = threadIdx.x; task < ntask; task += blockDim.x) {
        int il = task >> 2;
        int hh = task & (HN - 1);
        int bi = members[off + il];
        const float* qp = qkv + ((long)t * BN + bi) * (3 * DN) + hh * HD;

        float q[HD];
#pragma unroll
        for (int c = 0; c < HD; ++c) q[c] = qp[c] * 0.125f;   // 1/sqrt(64)

        // pass 1: row max
        float mx = -3.0e38f;
        for (int jl = 0; jl < G; ++jl) {
            int bj = members[off + jl];
            const float* kp = qkv + ((long)t * BN + bj) * (3 * DN) + DN + hh * HD;
            float s = 0.f;
#pragma unroll
            for (int c = 0; c < HD; ++c) s += q[c] * kp[c];
            mx = fmaxf(mx, s);
        }
        // pass 2: exp-sum + weighted V accumulate
        float o[HD];
#pragma unroll
        for (int c = 0; c < HD; ++c) o[c] = 0.f;
        float den = 0.f;
        for (int jl = 0; jl < G; ++jl) {
            int bj = members[off + jl];
            const float* kp = qkv + ((long)t * BN + bj) * (3 * DN) + DN + hh * HD;
            const float* vp = kp + DN;
            float s = 0.f;
#pragma unroll
            for (int c = 0; c < HD; ++c) s += q[c] * kp[c];
            float w = __expf(s - mx);
            den += w;
#pragma unroll
            for (int c = 0; c < HD; ++c) o[c] += w * vp[c];
        }
        float inv = 1.f / den;
        __bf16* op = attn_bf + ((long)t * BN + bi) * DN + hh * HD;
#pragma unroll
        for (int c = 0; c < HD; ++c) op[c] = (__bf16)(o[c] * inv);
    }
}

// ---------------------------------------------------------------------------
// K6: gate2 dot + sigmoid, gated residual update, LayerNorm. Block per (b,t).
// ---------------------------------------------------------------------------
__device__ inline float tc_block_sum(float v, float* sbuf) {
    int lane = threadIdx.x & 31;
    int wv   = threadIdx.x >> 5;
#pragma unroll
    for (int o = 16; o > 0; o >>= 1) v += __shfl_down(v, o, 32);
    if (lane == 0) sbuf[wv] = v;
    __syncthreads();
    if (wv == 0) {
        float tt = (lane < 8) ? sbuf[lane] : 0.f;
#pragma unroll
        for (int o = 4; o > 0; o >>= 1) tt += __shfl_down(tt, o, 32);
        if (lane == 0) sbuf[0] = tt;
    }
    __syncthreads();
    float r = sbuf[0];
    __syncthreads();
    return r;
}

__global__ void __launch_bounds__(DN)
tc_finalize(const float* __restrict__ x,
            const float* __restrict__ hbuf,
            const float* __restrict__ w2,
            const float* __restrict__ b2,
            const float* __restrict__ oproj,
            const int* __restrict__ multi,
            const float* __restrict__ gamma,
            const float* __restrict__ beta,
            float* __restrict__ out) {
    __shared__ float sbuf[8];
    int t = blockIdx.x & (TN - 1);
    int b = blockIdx.x >> 2;
    int e = threadIdx.x;

    float hv   = hbuf[((long)t * BN + b) * DN + e];
    float graw = tc_block_sum(hv * w2[t * DN + e], sbuf);
    float gate = 1.f / (1.f + __expf(-(graw + b2[t])));

    float xs = x[((long)b * TN + t) * DN + e];
    float ov = oproj[((long)t * BN + b) * DN + e];
    float u  = multi[b] ? (gate * xs + (1.f - gate) * ov) : xs;
    float y  = u + xs;

    float mu  = tc_block_sum(y, sbuf) * (1.f / DN);
    float msq = tc_block_sum(y * y, sbuf) * (1.f / DN);
    float var = msq - mu * mu;
    out[((long)b * TN + t) * DN + e] =
        (y - mu) * rsqrtf(var + 1e-5f) * gamma[e] + beta[e];
}

// ---------------------------------------------------------------------------
// Host launcher
// ---------------------------------------------------------------------------
extern "C" void kernel_launch(void* const* d_in, const int* in_sizes, int n_in,
                              void* d_out, int out_size, void* d_ws, size_t ws_size,
                              hipStream_t stream) {
    const float* x     = (const float*)d_in[0];
    const int*   ids   = (const int*)  d_in[1];
    const float* in_w  = (const float*)d_in[2];
    const float* in_b  = (const float*)d_in[3];
    const float* out_w = (const float*)d_in[4];
    const float* out_b = (const float*)d_in[5];
    const float* g_w1  = (const float*)d_in[6];
    const float* g_b1  = (const float*)d_in[7];
    const float* g_w2  = (const float*)d_in[8];
    const float* g_b2  = (const float*)d_in[9];
    const float* gamma = (const float*)d_in[10];
    const float* beta  = (const float*)d_in[11];
    float* out = (float*)d_out;

    // workspace carve-out (256B aligned)
    char* ws = (char*)d_ws;
    auto carve = [&](size_t bytes) {
        char* p = ws;
        ws += (bytes + 255) & ~(size_t)255;
        return p;
    };
    int*    counts  = (int*)   carve(NIMG * sizeof(int));
    int*    offs    = (int*)   carve(NIMG * sizeof(int));
    int*    members = (int*)   carve(BN * sizeof(int));
    int*    multi   = (int*)   carve(BN * sizeof(int));
    __bf16* xsb     = (__bf16*)carve((size_t)TN * BN * DN * 2);
    __bf16* cc      = (__bf16*)carve((size_t)TN * BN * 2 * DN * 2);
    __bf16* wqkv_b  = (__bf16*)carve((size_t)TN * 3 * DN * DN * 2);
    __bf16* wout_b  = (__bf16*)carve((size_t)TN * DN * DN * 2);
    __bf16* w1_b    = (__bf16*)carve((size_t)TN * DN * 2 * DN * 2);
    float*  qkv     = (float*) carve((size_t)TN * BN * 3 * DN * 4);
    __bf16* attn_bf = (__bf16*)carve((size_t)TN * BN * DN * 2);
    float*  oproj   = (float*) carve((size_t)TN * BN * DN * 4);
    float*  hbuf    = (float*) carve((size_t)TN * BN * DN * 4);

    // K0: groups
    tc_build_groups<<<1, 256, 0, stream>>>(ids, counts, offs, members, multi);

    // K1: precision conversion
    {
        long n = (long)TN * BN * DN;
        tc_convert_x<<<(int)((n + 255) / 256), 256, 0, stream>>>(x, xsb, cc);
        long nw1 = (long)TN * 3 * DN * DN;
        tc_convert_flat<<<(int)((nw1 + 255) / 256), 256, 0, stream>>>(in_w, wqkv_b, nw1);
        long nw2 = (long)TN * DN * DN;
        tc_convert_flat<<<(int)((nw2 + 255) / 256), 256, 0, stream>>>(out_w, wout_b, nw2);
        long nw3 = (long)TN * DN * 2 * DN;
        tc_convert_flat<<<(int)((nw3 + 255) / 256), 256, 0, stream>>>(g_w1, w1_b, nw3);
    }

    // K2: QKV projection  qkv[t] = xsb[t] (2048x256) * in_w[t]^T (768x256) + b
    {
        int blocks = TN * (BN / 256) * ((3 * DN) / 64);   // 4*8*12 = 384
        tc_gemm_wmma_bf16<DN><<<blocks, 256, 0, stream>>>(
            xsb, (long)BN * DN, wqkv_b, (long)3 * DN * DN,
            in_b, 3 * DN,
            qkv, (long)BN * 3 * DN,
            (__bf16*)nullptr, 0, 0,
            BN, 3 * DN, 0);
    }

    // K3: grouped attention -> attn_bf [T,B,D]
    tc_attention<<<NIMG * TN, 256, 0, stream>>>(qkv, counts, offs, members, attn_bf);

    // K4: out_proj  oproj[t] = attn[t] * out_w[t]^T + b ; bf16 mirror into cc[:,256:]
    {
        int blocks = TN * (BN / 256) * (DN / 64);         // 4*8*4 = 128
        tc_gemm_wmma_bf16<DN><<<blocks, 256, 0, stream>>>(
            attn_bf, (long)BN * DN, wout_b, (long)DN * DN,
            out_b, DN,
            oproj, (long)BN * DN,
            cc + DN, (long)BN * 2 * DN, 2 * DN,
            BN, DN, 0);
    }

    // K5: gate layer 1  hbuf[t] = relu(cc[t] (2048x512) * g_w1[t]^T (256x512) + b1)
    {
        int blocks = TN * (BN / 256) * (DN / 64);         // 128
        tc_gemm_wmma_bf16<2 * DN><<<blocks, 256, 0, stream>>>(
            cc, (long)BN * 2 * DN, w1_b, (long)DN * 2 * DN,
            g_b1, DN,
            hbuf, (long)BN * DN,
            (__bf16*)nullptr, 0, 0,
            BN, DN, 1);
    }

    // K6: gate2 + sigmoid + gated residual + LayerNorm
    tc_finalize<<<BN * TN, DN, 0, stream>>>(x, hbuf, g_w2, g_b2, oproj, multi,
                                            gamma, beta, out);
}
